// TA_LSTM_44667659878587
// MI455X (gfx1250) — matrix operand order repaired and checked by reading, not compiled, not run on hardware
//
#include <hip/hip_runtime.h>
#include <hip/hip_bf16.h>

// ---------------------------------------------------------------------------
// TA-LSTM for MI455X (gfx1250): WMMA bf16 GEMMs (LDS double-buffered, async
// global->LDS staging) + single-kernel LSTM scan.
// B=4096, S=128, D=64, H=256, IN=8192, OUT=1
// ---------------------------------------------------------------------------

#define B_   4096
#define S_   128
#define D_   64
#define H_   256
#define IN_  8192
#define H4_  1024
#define SH_  32768   // S*H

typedef __attribute__((ext_vector_type(16))) __bf16 bf16x16;
typedef __attribute__((ext_vector_type(8)))  float  floatx8;

struct __align__(16) u128 { unsigned int a, b, c, d; };

// --- fragment loaders (CDNA5 WMMA 16x16x32 bf16 layouts, ISA 7.12.2) -------
// A (16x32, MxK): lane L<16 holds M=L, K in {8h..8h+7, 16+8h..16+8h+7}, h=L>>4
__device__ __forceinline__ bf16x16 load_frag_a(const __bf16* A, int lda, int m0, int k0) {
    int lane = threadIdx.x & 31;
    const __bf16* p = A + (size_t)(m0 + (lane & 15)) * lda + k0 + ((lane >> 4) << 3);
    union { bf16x16 v; u128 q[2]; } u;
    u.q[0] = *(const u128*)(p);
    u.q[1] = *(const u128*)(p + 16);
    return u.v;
}

// B (32x16, KxN) from row-major NxK weight (TN GEMM): lane holds N=lane&15,
// K in {8h..8h+7, 16+8h..16+8h+7}, h=lane>>4 (same as A).
__device__ __forceinline__ bf16x16 load_frag_b(const __bf16* W, int ldw, int n0, int k0) {
    int lane = threadIdx.x & 31;
    const __bf16* p = W + (size_t)(n0 + (lane & 15)) * ldw + k0 + ((lane >> 4) << 3);
    union { bf16x16 v; u128 q[2]; } u;
    u.q[0] = *(const u128*)(p);
    u.q[1] = *(const u128*)(p + 16);
    return u.v;
}

__device__ __forceinline__ floatx8 wmma_bf16(bf16x16 a, bf16x16 b, floatx8 c) {
    return __builtin_amdgcn_wmma_f32_16x16x32_bf16(false, a, false, b, (short)0, c, false, false);
}

__device__ __forceinline__ float sigmoidf_(float x) { return 1.0f / (1.0f + __expf(-x)); }

// --- async global->LDS staging (CDNA5 GLOBAL_LOAD_ASYNC_TO_LDS_B128) -------
// Builtin signature (from toolchain diagnostic): params are v4i* in AS(1)/AS(3).
typedef __attribute__((vector_size(16))) int v4i_;
typedef __attribute__((address_space(1))) v4i_ as1_v4i;
typedef __attribute__((address_space(3))) v4i_ as3_v4i;

__device__ __forceinline__ void stage16(const __bf16* g, __bf16* l) {
#if __has_builtin(__builtin_amdgcn_global_load_async_to_lds_b128)
    __builtin_amdgcn_global_load_async_to_lds_b128(
        (as1_v4i*)g, (as3_v4i*)l, 0, 0);
#else
    *(u128*)l = *(const u128*)g;
#endif
}
__device__ __forceinline__ void stage_wait() {
#if __has_builtin(__builtin_amdgcn_global_load_async_to_lds_b128) && \
    __has_builtin(__builtin_amdgcn_s_wait_asynccnt)
    __builtin_amdgcn_s_wait_asynccnt(0);
#endif
}

// --- BatchNorm statistics (column-wise over 4096 rows) ----------------------
__global__ __launch_bounds__(256) void bn_stats_kernel(
    const float* __restrict__ x, float* __restrict__ mean, float* __restrict__ rstd) {
    int j = blockIdx.x * 256 + threadIdx.x;            // column 0..IN-1
    float s = 0.f, sq = 0.f;
    for (int r = 0; r < B_; ++r) {
        float v = x[(size_t)r * IN_ + j];
        s += v; sq += v * v;
    }
    float m   = s * (1.0f / (float)B_);
    float var = sq * (1.0f / (float)B_) - m * m;
    mean[j] = m;
    rstd[j] = rsqrtf(var + 1e-5f);
}

// --- normalize + cast to bf16 ----------------------------------------------
__global__ __launch_bounds__(256) void bn_norm_kernel(
    const float* __restrict__ x, const float* __restrict__ mean,
    const float* __restrict__ rstd, const float* __restrict__ gamma,
    const float* __restrict__ beta, __bf16* __restrict__ xn) {
    size_t i = (size_t)blockIdx.x * 256 + threadIdx.x;
    int j = (int)(i & (IN_ - 1));
    float v = (x[i] - mean[j]) * rstd[j] * gamma[j] + beta[j];
    xn[i] = (__bf16)v;
}

// --- generic f32 -> bf16 cast ----------------------------------------------
__global__ __launch_bounds__(256) void cast_bf16_kernel(
    const float* __restrict__ in, __bf16* __restrict__ out, size_t n) {
    size_t i = (size_t)blockIdx.x * 256 + threadIdx.x;
    if (i < n) out[i] = (__bf16)in[i];
}

// --- TN GEMM: C[M,N] = A[M,K] * B[N,K]^T, bf16 in, fp32 acc -----------------
// Block tile 128x128, k-step 32, double-buffered LDS staging (async copies).
// 8 waves in 4x2; wave tile 32x64 = 2x4 WMMA tiles (8 accumulators).
#define GEMM_BM 128
#define GEMM_BN 128
#define GEMM_BK 32
__global__ __launch_bounds__(256) void gemm_bf16_tn_kernel(
    const __bf16* __restrict__ A, const __bf16* __restrict__ Bm,
    float* __restrict__ Cf, __bf16* __restrict__ Cb,
    int M, int N, int K, int store_bf16) {
    __shared__ __bf16 Abuf[2][GEMM_BM * GEMM_BK];   // 8 KB each
    __shared__ __bf16 Bbuf[2][GEMM_BN * GEMM_BK];   // 8 KB each
    int tid = threadIdx.x;
    int w   = tid >> 5;
    int mw  = (w & 3) * 32;      // wave-local m base (4 wave rows)
    int nw  = (w >> 2) * 64;     // wave-local n base (2 wave cols)
    int blockM = blockIdx.y * GEMM_BM;
    int blockN = blockIdx.x * GEMM_BN;

    // tile-copy mapping: thread t stages one 16-B chunk (8 bf16) of one row of A and of B
    int crow = tid >> 1;                 // 0..127
    int ccol = (tid & 1) << 4;           // 0 or 16
    const __bf16* gA = A  + (size_t)(blockM + crow) * K + ccol;
    const __bf16* gB = Bm + (size_t)(blockN + crow) * K + ccol;
    int loff = crow * GEMM_BK + ccol;

    // prologue: stage k-tile 0 into buffer 0
    stage16(gA, &Abuf[0][loff]);
    stage16(gB, &Bbuf[0][loff]);

    floatx8 acc[2][4] = {};
    int nk = K / GEMM_BK;
    for (int kt = 0; kt < nk; ++kt) {
        int cur = kt & 1, nxt = cur ^ 1;
        stage_wait();          // my async copies into buf[cur] have landed
        __syncthreads();       // everyone's copies landed; prev reads done
        if (kt + 1 < nk) {     // stage next k-tile while computing this one
            stage16(gA + (size_t)(kt + 1) * GEMM_BK, &Abuf[nxt][loff]);
            stage16(gB + (size_t)(kt + 1) * GEMM_BK, &Bbuf[nxt][loff]);
        }
        bf16x16 a0 = load_frag_a(Abuf[cur], GEMM_BK, mw,      0);
        bf16x16 a1 = load_frag_a(Abuf[cur], GEMM_BK, mw + 16, 0);
        #pragma unroll
        for (int j = 0; j < 4; ++j) {
            bf16x16 bj = load_frag_b(Bbuf[cur], GEMM_BK, nw + j * 16, 0);
            acc[0][j] = wmma_bf16(a0, bj, acc[0][j]);
            acc[1][j] = wmma_bf16(a1, bj, acc[1][j]);
        }
    }

    int lane = tid & 31;
    int nn = lane & 15, mb = (lane >> 4) << 3;
    #pragma unroll
    for (int i = 0; i < 2; ++i)
        #pragma unroll
        for (int j = 0; j < 4; ++j) {
            int mt = blockM + mw + i * 16 + mb;
            int nt = blockN + nw + j * 16 + nn;
            #pragma unroll
            for (int r = 0; r < 8; ++r) {
                size_t idx = (size_t)(mt + r) * N + nt;
                if (store_bf16) Cb[idx] = (__bf16)acc[i][j][r];
                else            Cf[idx] = acc[i][j][r];
            }
        }
}

// --- LSTM scan: one workgroup = 16 batch rows, all 128 timesteps ------------
// gates[16,1024] = proj_s[16,64] @ Wih^T + h[16,256] @ Whh^T + bias
// Each of 8 waves owns 32 hidden units (2 N-tiles x 4 gate groups = 8 accums).
// Cell state c stays in VGPRs; h lives in LDS as bf16; h_s streamed to hs.
__global__ __launch_bounds__(256) void lstm_scan_kernel(
    const __bf16* __restrict__ projb,   // [B, IN] bf16
    const __bf16* __restrict__ Wih,     // [1024, 64]
    const __bf16* __restrict__ Whh,     // [1024, 256]
    const float*  __restrict__ bih, const float* __restrict__ bhh,
    __bf16* __restrict__ hs) {          // [B, S*H]
    __shared__ __bf16 h_lds[16 * H_];
    int w    = threadIdx.x >> 5;
    int lane = threadIdx.x & 31;
    int b0   = blockIdx.x * 16;
    int hb   = w * 32;                  // hidden-column base for this wave

    for (int i = threadIdx.x; i < 16 * H_; i += 256) h_lds[i] = (__bf16)0.0f;
    __syncthreads();

    float biasv[4][2];
    #pragma unroll
    for (int g = 0; g < 4; ++g)
        #pragma unroll
        for (int t = 0; t < 2; ++t) {
            int n = g * H_ + hb + t * 16 + (lane & 15);
            biasv[g][t] = bih[n] + bhh[n];
        }

    floatx8 c[2] = {};
    for (int s = 0; s < S_; ++s) {
        floatx8 acc[4][2] = {};
        // x-side gates: K = 64 (columns s*64.. of proj), weights Wih
        for (int kk = 0; kk < D_; kk += 32) {
            bf16x16 a = load_frag_a(projb, IN_, b0, s * D_ + kk);
            #pragma unroll
            for (int g = 0; g < 4; ++g)
                #pragma unroll
                for (int t = 0; t < 2; ++t) {
                    bf16x16 bb = load_frag_b(Wih, D_, g * H_ + hb + t * 16, kk);
                    acc[g][t] = wmma_bf16(a, bb, acc[g][t]);
                }
        }
        // h-side gates: K = 256 from LDS h, weights Whh
        for (int kk = 0; kk < H_; kk += 32) {
            bf16x16 a = load_frag_a(h_lds, H_, 0, kk);
            #pragma unroll
            for (int g = 0; g < 4; ++g)
                #pragma unroll
                for (int t = 0; t < 2; ++t) {
                    bf16x16 bb = load_frag_b(Whh, H_, g * H_ + hb + t * 16, kk);
                    acc[g][t] = wmma_bf16(a, bb, acc[g][t]);
                }
        }
        __syncthreads();   // all waves finished reading h_lds
        #pragma unroll
        for (int t = 0; t < 2; ++t) {
            #pragma unroll
            for (int r = 0; r < 8; ++r) {
                float iv = sigmoidf_(acc[0][t][r] + biasv[0][t]);
                float fv = sigmoidf_(acc[1][t][r] + biasv[1][t]);
                float gv = tanhf   (acc[2][t][r] + biasv[2][t]);
                float ov = sigmoidf_(acc[3][t][r] + biasv[3][t]);
                float cv = fv * c[t][r] + iv * gv;
                c[t][r] = cv;
                float hv = ov * tanhf(cv);
                int m = ((lane >> 4) << 3) + r;            // 0..15 (C/D layout)
                int n = hb + t * 16 + (lane & 15);
                __bf16 hvb = (__bf16)hv;
                h_lds[m * H_ + n] = hvb;
                hs[(size_t)(b0 + m) * SH_ + (size_t)s * H_ + n] = hvb;
            }
        }
        __syncthreads();   // h updates visible before next step's reads
    }
}

// --- relu + softmax over S=128, in place (logits -> beta_t) -----------------
__global__ __launch_bounds__(128) void attn_softmax_kernel(
    float* __restrict__ logits, const float* __restrict__ bta) {
    __shared__ float red[128];
    int b = blockIdx.x, j = threadIdx.x;
    float v = fmaxf(logits[(size_t)b * S_ + j] + bta[j], 0.0f);
    red[j] = v; __syncthreads();
    for (int st = 64; st > 0; st >>= 1) {
        if (j < st) red[j] = fmaxf(red[j], red[j + st]);
        __syncthreads();
    }
    float mx = red[0]; __syncthreads();
    float e = __expf(v - mx);
    red[j] = e; __syncthreads();
    for (int st = 64; st > 0; st >>= 1) {
        if (j < st) red[j] += red[j + st];
        __syncthreads();
    }
    logits[(size_t)b * S_ + j] = e / red[0];
}

// --- pooled[b,h] = sum_s hs[b,s,h] * beta_t[b,s] ----------------------------
__global__ __launch_bounds__(256) void attn_pool_kernel(
    const __bf16* __restrict__ hs, const float* __restrict__ beta_t,
    float* __restrict__ pooled) {
    int idx = blockIdx.x * 256 + threadIdx.x;          // b*H + h
    int b = idx >> 8, h = idx & (H_ - 1);
    const __bf16* p = hs + (size_t)b * SH_ + h;
    const float*  wt = beta_t + (size_t)b * S_;
    float acc = 0.f;
    #pragma unroll 4
    for (int s = 0; s < S_; ++s) acc += (float)p[(size_t)s * H_] * wt[s];
    pooled[idx] = acc;
}

// --- out[b] = pooled[b,:] . W_out ------------------------------------------
__global__ __launch_bounds__(256) void out_proj_kernel(
    const float* __restrict__ pooled, const float* __restrict__ Wout,
    float* __restrict__ out) {
    __shared__ float red[256];
    int b = blockIdx.x, h = threadIdx.x;
    red[h] = pooled[(size_t)b * H_ + h] * Wout[h];
    __syncthreads();
    for (int st = 128; st > 0; st >>= 1) {
        if (h < st) red[h] += red[h + st];
        __syncthreads();
    }
    if (h == 0) out[b] = red[0];
}

// ---------------------------------------------------------------------------
extern "C" void kernel_launch(void* const* d_in, const int* in_sizes, int n_in,
                              void* d_out, int out_size, void* d_ws, size_t ws_size,
                              hipStream_t stream) {
    const float* x     = (const float*)d_in[0];
    const float* gamma = (const float*)d_in[1];
    const float* beta  = (const float*)d_in[2];
    const float* W_in  = (const float*)d_in[3];
    const float* W_ih  = (const float*)d_in[4];
    const float* b_ih  = (const float*)d_in[5];
    const float* W_hh  = (const float*)d_in[6];
    const float* b_hh  = (const float*)d_in[7];
    const float* W_ta  = (const float*)d_in[8];
    const float* b_ta  = (const float*)d_in[9];
    const float* W_out = (const float*)d_in[10];
    float* out = (float*)d_out;

    char* ws = (char*)d_ws;
    size_t off = 0;
    auto alloc = [&](size_t bytes) -> void* {
        off = (off + 255) & ~(size_t)255;
        void* p = ws + off;
        off += bytes;
        return p;
    };

    float*  mean  = (float*) alloc((size_t)IN_ * 4);
    float*  rstd  = (float*) alloc((size_t)IN_ * 4);
    __bf16* xn    = (__bf16*)alloc((size_t)B_ * IN_ * 2);        //  64 MB
    __bf16* Winb  = (__bf16*)alloc((size_t)IN_ * IN_ * 2);       // 128 MB (L2-resident)
    __bf16* projb = (__bf16*)alloc((size_t)B_ * IN_ * 2);        //  64 MB
    __bf16* Wihb  = (__bf16*)alloc((size_t)H4_ * D_ * 2);
    __bf16* Whhb  = (__bf16*)alloc((size_t)H4_ * H_ * 2);
    __bf16* Wtab  = (__bf16*)alloc((size_t)S_ * SH_ * 2);        //   8 MB
    __bf16* hsb   = (__bf16*)alloc((size_t)B_ * SH_ * 2);        // 256 MB
    float*  logit = (float*) alloc((size_t)B_ * S_ * 4);
    float*  pool  = (float*) alloc((size_t)B_ * H_ * 4);

    // 1) BatchNorm
    bn_stats_kernel<<<IN_ / 256, 256, 0, stream>>>(x, mean, rstd);
    bn_norm_kernel<<<(int)(((size_t)B_ * IN_) / 256), 256, 0, stream>>>(
        x, mean, rstd, gamma, beta, xn);

    // 2) weight casts to bf16
    cast_bf16_kernel<<<(int)(((size_t)IN_ * IN_ + 255) / 256), 256, 0, stream>>>(
        W_in, Winb, (size_t)IN_ * IN_);
    cast_bf16_kernel<<<(H4_ * D_ + 255) / 256, 256, 0, stream>>>(
        W_ih, Wihb, (size_t)H4_ * D_);
    cast_bf16_kernel<<<(H4_ * H_ + 255) / 256, 256, 0, stream>>>(
        W_hh, Whhb, (size_t)H4_ * H_);
    cast_bf16_kernel<<<(int)(((size_t)S_ * SH_ + 255) / 256), 256, 0, stream>>>(
        W_ta, Wtab, (size_t)S_ * SH_);

    // 3) proj = xn @ W_in^T   (M=4096, N=8192, K=8192) -> bf16
    gemm_bf16_tn_kernel<<<dim3(IN_ / GEMM_BN, B_ / GEMM_BM), 256, 0, stream>>>(
        xn, Winb, nullptr, projb, B_, IN_, IN_, 1);

    // 4) full LSTM scan (per-batch-row recurrence, 16 rows/workgroup)
    lstm_scan_kernel<<<B_ / 16, 256, 0, stream>>>(projb, Wihb, Whhb, b_ih, b_hh, hsb);

    // 5) attention logits = total_ht @ W_ta^T  (M=4096, N=128, K=32768) -> f32
    gemm_bf16_tn_kernel<<<dim3(S_ / GEMM_BN, B_ / GEMM_BM), 256, 0, stream>>>(
        hsb, Wtab, logit, nullptr, B_, S_, SH_, 0);

    // 6) relu + softmax (in place: logits -> beta_t)
    attn_softmax_kernel<<<B_, 128, 0, stream>>>(logit, b_ta);

    // 7) pooled = sum_s hs * beta
    attn_pool_kernel<<<(B_ * H_) / 256, 256, 0, stream>>>(hsb, logit, pool);

    // 8) out = pooled @ W_out^T
    out_proj_kernel<<<B_, 256, 0, stream>>>(pool, W_out, out);
}
